// TransformerLayer_80719615361628
// MI455X (gfx1250) — matrix operand order repaired and checked by reading, not compiled, not run on hardware
//
#include <hip/hip_runtime.h>

// ---------------------------------------------------------------------------
// Transformer encoder layer forward for MI455X (gfx1250, wave32, WMMA).
// Matmuls: v_wmma_f32_16x16x32_f16 (f16 in, f32 accum).
// Data movement: global_load_async_to_lds_b128 (ASYNCcnt) + double-buffered
// LDS so tile staging overlaps WMMA compute.
// ---------------------------------------------------------------------------

#define D_MODEL   1024
#define N_HEADS   16
#define HEAD_DIM  64
#define FF_DIM    4096
#define BATCH     2
#define SEQ       2048
#define ROWS      (BATCH * SEQ)   // 4096 token rows
#define TILE_ELEMS (128 * 32)     // one 128x32 f16 GEMM tile

typedef __attribute__((ext_vector_type(16))) _Float16 v16h;
typedef __attribute__((ext_vector_type(8)))  _Float16 v8h;
typedef __attribute__((ext_vector_type(4)))  _Float16 v4h;
typedef __attribute__((ext_vector_type(8)))  float    v8f;
typedef __attribute__((ext_vector_type(4)))  float    v4f;

union F16x16 { v16h v; uint32_t u[8]; };

__device__ __forceinline__ v8f wmma_f32_f16(v16h a, v16h b, v8f c) {
  // D = A(16x32 f16) * B(32x16 f16) + C(16x16 f32)
  return __builtin_amdgcn_wmma_f32_16x16x32_f16(
      /*neg_a=*/false, a, /*neg_b=*/false, b,
      /*c_mod=*/(short)0, c, /*reuse_a=*/false, /*reuse_b=*/false);
}

// Async 16B global -> LDS copy (per-lane addresses), tracked by ASYNCcnt.
// LDS operand wants the LDS byte offset: low 32 bits of the flat address.
__device__ __forceinline__ void async_b128(void* lds, const void* g) {
  asm volatile("global_load_async_to_lds_b128 %0, %1, off"
               :: "v"((uint32_t)(uintptr_t)lds), "v"(g)
               : "memory");
}
__device__ __forceinline__ void wait_async0() {
#if __has_builtin(__builtin_amdgcn_s_wait_asynccnt)
  __builtin_amdgcn_s_wait_asynccnt(0);
#else
  asm volatile("s_wait_asynccnt 0x0" ::: "memory");
#endif
}

// A fragment (16x32 f16, MxK) from LDS tile stored row-major [m][k].
// lanes 0-15 row M=lane: VGPR0-3 => K 0..7, VGPR4-7 => K 16..23;
// lanes 16-31 add +8 to K within each 16-wide group.
__device__ __forceinline__ v16h load_a_frag(const _Float16* base, int ldk,
                                            int row_off, int kb, int lane) {
  F16x16 r;
  const int m = row_off + (lane & 15);
  const uint32_t* p = (const uint32_t*)(base + (size_t)m * ldk);
  const int kd = (kb + ((lane & 16) ? 8 : 0)) >> 1;  // dword base
#pragma unroll
  for (int vv = 0; vv < 8; ++vv) {
    const int koff = (vv < 4) ? vv : (8 + (vv - 4));
    r.u[vv] = p[kd + koff];
  }
  return r.v;
}

// B fragment (32x16 f16, KxN) from LDS tile stored [n][k] (K contiguous).
// lanes 0-15 column N=lane: VGPR0-3 => K 0..7, VGPR4-7 => K 16..23;
// lanes 16-31 add +8 to K within each 16-wide group.
__device__ __forceinline__ v16h load_b_frag(const _Float16* base, int ldk,
                                            int n_off, int kb, int lane) {
  F16x16 r;
  const int n = n_off + (lane & 15);
  const uint32_t* p = (const uint32_t*)(base + (size_t)n * ldk) +
                      ((kb + ((lane & 16) ? 8 : 0)) >> 1);
#pragma unroll
  for (int vv = 0; vv < 8; ++vv) r.u[vv] = p[(vv < 4) ? vv : (8 + (vv - 4))];
  return r.v;
}

// ---------------------------------------------------------------------------
// f32 -> f16 weight conversion (element counts are multiples of 1024).
// ---------------------------------------------------------------------------
__global__ __launch_bounds__(256) void cvt_f32_f16_kernel(
    const float* __restrict__ in, _Float16* __restrict__ outp) {
  const int i = (blockIdx.x * 256 + threadIdx.x) * 4;
  const v4f a = *(const v4f*)(in + i);
  v4h h;
  h[0] = (_Float16)a[0]; h[1] = (_Float16)a[1];
  h[2] = (_Float16)a[2]; h[3] = (_Float16)a[3];
  *(v4h*)(outp + i) = h;
}

// ---------------------------------------------------------------------------
// LayerNorm: one 256-thread block per row (D_MODEL=1024 -> 4 elems/thread),
// writes f16 activations (WMMA A operand).
// ---------------------------------------------------------------------------
__global__ __launch_bounds__(256) void ln_kernel(
    const float* __restrict__ x, const float* __restrict__ g,
    const float* __restrict__ b, _Float16* __restrict__ out) {
  const int row = blockIdx.x;
  const float* xr = x + (size_t)row * D_MODEL;
  const int t = threadIdx.x;
  float vals[4];
  float s = 0.f;
#pragma unroll
  for (int i = 0; i < 4; ++i) { vals[i] = xr[t + i * 256]; s += vals[i]; }
#pragma unroll
  for (int m = 16; m > 0; m >>= 1) s += __shfl_xor(s, m, 32);
  __shared__ float red[8];
  if ((t & 31) == 0) red[t >> 5] = s;
  __syncthreads();
  float tot = 0.f;
#pragma unroll
  for (int i = 0; i < 8; ++i) tot += red[i];
  const float mu = tot * (1.0f / (float)D_MODEL);
  float s2 = 0.f;
#pragma unroll
  for (int i = 0; i < 4; ++i) { const float d = vals[i] - mu; s2 += d * d; }
#pragma unroll
  for (int m = 16; m > 0; m >>= 1) s2 += __shfl_xor(s2, m, 32);
  __syncthreads();
  if ((t & 31) == 0) red[t >> 5] = s2;
  __syncthreads();
  float tot2 = 0.f;
#pragma unroll
  for (int i = 0; i < 8; ++i) tot2 += red[i];
  const float rstd = rsqrtf(tot2 * (1.0f / (float)D_MODEL) + 1e-5f);
  _Float16* orow = out + (size_t)row * D_MODEL;
#pragma unroll
  for (int i = 0; i < 4; ++i) {
    const int idx = t + i * 256;
    orow[idx] = (_Float16)((vals[i] - mu) * rstd * g[idx] + b[idx]);
  }
}

// ---------------------------------------------------------------------------
// GEMM main loop: out[m][n] = sum_k A[m][k] * Wh[n][k]  (out = A @ W^T)
// Block tile 128x128, K-step 32; 8 waves in 2(M)x4(N) grid, each wave owns a
// 64x32 patch = 4x2 WMMA accumulators. Both operands f16 and K-contiguous ->
// async-copied straight into double-buffered LDS; tile (k+1) streams while
// tile k feeds the WMMAs. One barrier + one s_wait_asynccnt per K-step.
// ---------------------------------------------------------------------------
__device__ __forceinline__ void copy_gemm_tiles(
    _Float16* Ad, _Float16* Bd, const _Float16* __restrict__ A,
    const _Float16* __restrict__ Wh, int K, int m0, int n0, int k0, int t) {
#pragma unroll
  for (int i = 0; i < 2; ++i) {
    const int lin = t + i * 256;
    const int row = lin >> 2;
    const int col = (lin & 3) << 3;
    async_b128(Ad + row * 32 + col, A + (size_t)(m0 + row) * K + k0 + col);
    async_b128(Bd + row * 32 + col, Wh + (size_t)(n0 + row) * K + k0 + col);
  }
}

__device__ __forceinline__ void gemm_mainloop(
    _Float16* As, _Float16* Bs, const _Float16* __restrict__ A,
    const _Float16* __restrict__ Wh, int K, int m0, int n0, int t,
    v8f acc[4][2]) {
  const int lane = t & 31;
  const int wave = t >> 5;
  const int wm = wave >> 2;   // 0..1
  const int wn = wave & 3;    // 0..3
  const int nk = K >> 5;
  copy_gemm_tiles(As, Bs, A, Wh, K, m0, n0, 0, t);
  wait_async0();
  __syncthreads();
  for (int ki = 0; ki < nk; ++ki) {
    const int cur = ki & 1;
    const _Float16* Ac = As + cur * TILE_ELEMS;
    const _Float16* Bc = Bs + cur * TILE_ELEMS;
    if (ki + 1 < nk)  // uniform (scalar) branch: stream next tile
      copy_gemm_tiles(As + (cur ^ 1) * TILE_ELEMS, Bs + (cur ^ 1) * TILE_ELEMS,
                      A, Wh, K, m0, n0, (ki + 1) << 5, t);
    v16h af[4], bf[2];
#pragma unroll
    for (int i = 0; i < 4; ++i)
      af[i] = load_a_frag(Ac, 32, wm * 64 + i * 16, 0, lane);
#pragma unroll
    for (int j = 0; j < 2; ++j)
      bf[j] = load_b_frag(Bc, 32, wn * 32 + j * 16, 0, lane);
#pragma unroll
    for (int i = 0; i < 4; ++i)
#pragma unroll
      for (int j = 0; j < 2; ++j)
        acc[i][j] = wmma_f32_f16(af[i], bf[j], acc[i][j]);
    if (ki + 1 < nk) { wait_async0(); __syncthreads(); }
  }
}

// ---------------------------------------------------------------------------
// QKV projection: grid.z selects {q,k,v}. Epilogue adds bias and scatters
// f16 into [b][h][s][hd] so attention reads contiguous HEAD_DIM vectors.
// ---------------------------------------------------------------------------
__global__ __launch_bounds__(256) void qkv_gemm_kernel(
    const _Float16* __restrict__ A,
    const _Float16* __restrict__ WqH, const float* __restrict__ bq,
    const _Float16* __restrict__ WkH, const float* __restrict__ bk,
    const _Float16* __restrict__ WvH, const float* __restrict__ bv,
    _Float16* __restrict__ qo, _Float16* __restrict__ ko,
    _Float16* __restrict__ vo) {
  __shared__ __align__(16) _Float16 As[2 * TILE_ELEMS];
  __shared__ __align__(16) _Float16 Bs[2 * TILE_ELEMS];
  const _Float16* W  = (blockIdx.z == 0) ? WqH : (blockIdx.z == 1) ? WkH : WvH;
  const float*    bb = (blockIdx.z == 0) ? bq  : (blockIdx.z == 1) ? bk  : bv;
  _Float16*    outp  = (blockIdx.z == 0) ? qo  : (blockIdx.z == 1) ? ko  : vo;
  const int t  = threadIdx.x;
  const int m0 = blockIdx.y * 128;
  const int n0 = blockIdx.x * 128;
  v8f acc[4][2];
  const v8f z = {};
#pragma unroll
  for (int i = 0; i < 4; ++i)
#pragma unroll
    for (int j = 0; j < 2; ++j) acc[i][j] = z;
  gemm_mainloop(As, Bs, A, W, D_MODEL, m0, n0, t, acc);
  const int lane = t & 31;
  const int wave = t >> 5;
  const int mbase = m0 + (wave >> 2) * 64 + ((lane & 16) ? 8 : 0);
  const int nbase = n0 + (wave & 3) * 32 + (lane & 15);
#pragma unroll
  for (int i = 0; i < 4; ++i)
#pragma unroll
    for (int j = 0; j < 2; ++j)
#pragma unroll
      for (int r = 0; r < 8; ++r) {
        const int mm = mbase + i * 16 + r;
        const int nn = nbase + j * 16;
        const float val = acc[i][j][r] + bb[nn];
        const int bi = mm >> 11;            // / SEQ
        const int si = mm & (SEQ - 1);
        const int h  = nn >> 6;             // / HEAD_DIM
        const int hd = nn & (HEAD_DIM - 1);
        outp[((size_t)(bi * N_HEADS + h) * SEQ + si) * HEAD_DIM + hd] =
            (_Float16)val;
      }
}

// ---------------------------------------------------------------------------
// Generic GEMM epilogues:
//   EPI==0: f32 out = acc + bias + resid   (o-proj, FFN2 — residual adds)
//   EPI==1: f16 out = gelu(acc + bias)     (FFN1, exact erf GELU)
// ---------------------------------------------------------------------------
template <int EPI>
__global__ __launch_bounds__(256) void gemm_kernel(
    const _Float16* __restrict__ A, const _Float16* __restrict__ Wh,
    const float* __restrict__ bias, const float* __restrict__ resid,
    float* __restrict__ outF, _Float16* __restrict__ outH, int N, int K) {
  __shared__ __align__(16) _Float16 As[2 * TILE_ELEMS];
  __shared__ __align__(16) _Float16 Bs[2 * TILE_ELEMS];
  const int t  = threadIdx.x;
  const int m0 = blockIdx.y * 128;
  const int n0 = blockIdx.x * 128;
  v8f acc[4][2];
  const v8f z = {};
#pragma unroll
  for (int i = 0; i < 4; ++i)
#pragma unroll
    for (int j = 0; j < 2; ++j) acc[i][j] = z;
  gemm_mainloop(As, Bs, A, Wh, K, m0, n0, t, acc);
  const int lane = t & 31;
  const int wave = t >> 5;
  const int mbase = m0 + (wave >> 2) * 64 + ((lane & 16) ? 8 : 0);
  const int nbase = n0 + (wave & 3) * 32 + (lane & 15);
#pragma unroll
  for (int i = 0; i < 4; ++i)
#pragma unroll
    for (int j = 0; j < 2; ++j)
#pragma unroll
      for (int r = 0; r < 8; ++r) {
        const int mm = mbase + i * 16 + r;
        const int nn = nbase + j * 16;
        const float v = acc[i][j][r] + bias[nn];
        if (EPI == 0) {
          outF[(size_t)mm * N + nn] = v + resid[(size_t)mm * N + nn];
        } else {
          const float gv = 0.5f * v * (1.0f + erff(v * 0.70710678118654752f));
          outH[(size_t)mm * N + nn] = (_Float16)gv;
        }
      }
}

// ---------------------------------------------------------------------------
// Flash-style causal attention. One block = 64 query rows of one (b,h);
// 4 waves x 16 rows. Streams 64-wide key tiles up to the causal bound with
// online softmax (row stats reduced with half-wave shuffles: a 16x16 C tile
// keeps rows in 16-lane halves under wave32). Q/K tiles use async copies;
// V goes through VGPRs since it is transposed into LDS.
// ---------------------------------------------------------------------------
__global__ __launch_bounds__(128) void attn_kernel(
    const _Float16* __restrict__ q, const _Float16* __restrict__ k,
    const _Float16* __restrict__ v, _Float16* __restrict__ ctx) {
  __shared__ __align__(16) _Float16 Qs[64 * 64];
  __shared__ __align__(16) _Float16 Ks[64 * 64];
  __shared__ __align__(16) _Float16 Vt[64 * 64];  // transposed: [hd][key]
  __shared__ __align__(16) _Float16 Ps[64 * 64];  // probs (per-wave rows)

  const int t = threadIdx.x;
  const int lane = t & 31;
  const int wave = t >> 5;               // 0..3 -> query rows wave*16..+15
  const int bh = blockIdx.y;             // b*N_HEADS + h
  const int q0 = blockIdx.x * 64;
  const _Float16* qp = q + (size_t)bh * SEQ * HEAD_DIM;
  const _Float16* kp = k + (size_t)bh * SEQ * HEAD_DIM;
  const _Float16* vp = v + (size_t)bh * SEQ * HEAD_DIM;

#pragma unroll
  for (int i = 0; i < 4; ++i) {          // 64x64 Q tile via async copy
    const int lin = t + i * 128;
    const int row = lin >> 3;
    const int col = (lin & 7) << 3;
    async_b128(Qs + row * 64 + col, qp + (size_t)(q0 + row) * HEAD_DIM + col);
  }
  wait_async0();
  __syncthreads();

  const v16h qf0 = load_a_frag(Qs, 64, wave * 16, 0, lane);
  const v16h qf1 = load_a_frag(Qs, 64, wave * 16, 32, lane);

  v8f o[4];
  { const v8f z = {}; o[0] = z; o[1] = z; o[2] = z; o[3] = z; }
  float mrow[8], lrow[8];
#pragma unroll
  for (int r = 0; r < 8; ++r) { mrow[r] = -1e30f; lrow[r] = 0.0f; }

  const int hi8   = (lane & 16) ? 8 : 0;
  const int col16 = lane & 15;
  const int qrow0 = q0 + wave * 16 + hi8;

  for (int kt = 0; kt < q0 + 64; kt += 64) {   // causal key-tile bound
    __syncthreads();
    v8h vreg[4];
#pragma unroll
    for (int i = 0; i < 4; ++i) {       // issue V loads + async K copies first
      const int lin = t + i * 128;
      const int row = lin >> 3;
      const int col = (lin & 7) << 3;
      vreg[i] = *(const v8h*)(vp + (size_t)(kt + row) * HEAD_DIM + col);
      async_b128(Ks + row * 64 + col, kp + (size_t)(kt + row) * HEAD_DIM + col);
    }
#pragma unroll
    for (int i = 0; i < 4; ++i) {       // transpose V into LDS
      const int lin = t + i * 128;
      const int row = lin >> 3;
      const int col = (lin & 7) << 3;
#pragma unroll
      for (int e = 0; e < 8; ++e) Vt[(col + e) * 64 + row] = vreg[i][e];
    }
    wait_async0();
    __syncthreads();

    // scores = Q @ K^T  (K tile row-major [key][hd] == B layout [n][k])
    v8f s[4];
#pragma unroll
    for (int j = 0; j < 4; ++j) {
      v8f c = {};
      c = wmma_f32_f16(qf0, load_b_frag(Ks, 64, j * 16, 0, lane), c);
      c = wmma_f32_f16(qf1, load_b_frag(Ks, 64, j * 16, 32, lane), c);
      s[j] = c;
    }

    // scale + causal mask + row max
    float rmax[8];
#pragma unroll
    for (int r = 0; r < 8; ++r) rmax[r] = -1e30f;
#pragma unroll
    for (int j = 0; j < 4; ++j) {
      const int ki = kt + j * 16 + col16;
#pragma unroll
      for (int r = 0; r < 8; ++r) {
        const float val = (ki <= qrow0 + r) ? (s[j][r] * 0.125f) : -1e30f;
        s[j][r] = val;
        rmax[r] = fmaxf(rmax[r], val);
      }
    }
#pragma unroll
    for (int r = 0; r < 8; ++r) {
#pragma unroll
      for (int m = 8; m > 0; m >>= 1)
        rmax[r] = fmaxf(rmax[r], __shfl_xor(rmax[r], m, 16));
      const float mnew  = fmaxf(mrow[r], rmax[r]);
      const float alpha = __expf(mrow[r] - mnew);
      mrow[r] = mnew;
      lrow[r] *= alpha;
#pragma unroll
      for (int j = 0; j < 4; ++j) o[j][r] *= alpha;
    }

    // P = exp(S - m); stash in LDS (own-wave rows only -> no barrier needed)
    float rsum[8];
#pragma unroll
    for (int r = 0; r < 8; ++r) rsum[r] = 0.0f;
#pragma unroll
    for (int j = 0; j < 4; ++j)
#pragma unroll
      for (int r = 0; r < 8; ++r) {
        const float p = __expf(s[j][r] - mrow[r]);
        rsum[r] += p;
        Ps[(wave * 16 + r + hi8) * 64 + j * 16 + col16] = (_Float16)p;
      }
#pragma unroll
    for (int r = 0; r < 8; ++r) {
#pragma unroll
      for (int m = 8; m > 0; m >>= 1) rsum[r] += __shfl_xor(rsum[r], m, 16);
      lrow[r] += rsum[r];
    }

    // O += P @ V  (Vt is [hd][key] == B layout [n][k])
    const v16h pf0 = load_a_frag(Ps, 64, wave * 16, 0, lane);
    const v16h pf1 = load_a_frag(Ps, 64, wave * 16, 32, lane);
#pragma unroll
    for (int j = 0; j < 4; ++j) {
      o[j] = wmma_f32_f16(pf0, load_b_frag(Vt, 64, j * 16, 0, lane), o[j]);
      o[j] = wmma_f32_f16(pf1, load_b_frag(Vt, 64, j * 16, 32, lane), o[j]);
    }
  }

  // ctx back to token-major [b*S][D] (f16) for the output projection GEMM
  const int b_idx = bh >> 4;
  const int h_idx = bh & (N_HEADS - 1);
#pragma unroll
  for (int j = 0; j < 4; ++j)
#pragma unroll
    for (int r = 0; r < 8; ++r) {
      const int srow = q0 + wave * 16 + hi8 + r;
      const float val = o[j][r] / lrow[r];
      ctx[((size_t)(b_idx * SEQ + srow)) * D_MODEL + h_idx * HEAD_DIM +
          j * 16 + col16] = (_Float16)val;
    }
}

// ---------------------------------------------------------------------------
extern "C" void kernel_launch(void* const* d_in, const int* in_sizes, int n_in,
                              void* d_out, int out_size, void* d_ws,
                              size_t ws_size, hipStream_t stream) {
  (void)in_sizes; (void)n_in; (void)out_size; (void)ws_size;
  const float* x   = (const float*)d_in[0];
  // d_in[1] = attn_mask (tril ones) -> causal mask applied analytically
  const float* Wq  = (const float*)d_in[2];
  const float* bq  = (const float*)d_in[3];
  const float* Wk  = (const float*)d_in[4];
  const float* bk  = (const float*)d_in[5];
  const float* Wv  = (const float*)d_in[6];
  const float* bv  = (const float*)d_in[7];
  const float* Wo  = (const float*)d_in[8];
  const float* bo  = (const float*)d_in[9];
  const float* W1  = (const float*)d_in[10];
  const float* b1  = (const float*)d_in[11];
  const float* W2  = (const float*)d_in[12];
  const float* b2  = (const float*)d_in[13];
  const float* g1  = (const float*)d_in[14];
  const float* be1 = (const float*)d_in[15];
  const float* g2  = (const float*)d_in[16];
  const float* be2 = (const float*)d_in[17];
  float* out = (float*)d_out;

  char* ws = (char*)d_ws;
  const size_t MB = 1u << 20;
  _Float16* xn  = (_Float16*)(ws);            // 8 MB (reused LN1/LN2)
  _Float16* qB  = (_Float16*)(ws + 8 * MB);
  _Float16* kB  = (_Float16*)(ws + 16 * MB);
  _Float16* vB  = (_Float16*)(ws + 24 * MB);
  _Float16* ctx = (_Float16*)(ws + 32 * MB);
  float*    x2  = (float*)   (ws + 40 * MB);  // 16 MB f32 residual
  _Float16* hff = (_Float16*)(ws + 8 * MB);   // 32 MB, overlaps dead q..ctx
  _Float16* WqH = (_Float16*)(ws + 56 * MB);  // 2 MB each
  _Float16* WkH = (_Float16*)(ws + 58 * MB);
  _Float16* WvH = (_Float16*)(ws + 60 * MB);
  _Float16* WoH = (_Float16*)(ws + 62 * MB);
  _Float16* W1H = (_Float16*)(ws + 64 * MB);  // 8 MB
  _Float16* W2H = (_Float16*)(ws + 72 * MB);  // 8 MB  -> 80 MB total

  // 0) one-time (per launch) f32 -> f16 weight conversion
  cvt_f32_f16_kernel<<<(D_MODEL * D_MODEL) / 1024, 256, 0, stream>>>(Wq, WqH);
  cvt_f32_f16_kernel<<<(D_MODEL * D_MODEL) / 1024, 256, 0, stream>>>(Wk, WkH);
  cvt_f32_f16_kernel<<<(D_MODEL * D_MODEL) / 1024, 256, 0, stream>>>(Wv, WvH);
  cvt_f32_f16_kernel<<<(D_MODEL * D_MODEL) / 1024, 256, 0, stream>>>(Wo, WoH);
  cvt_f32_f16_kernel<<<(FF_DIM * D_MODEL) / 1024, 256, 0, stream>>>(W1, W1H);
  cvt_f32_f16_kernel<<<(FF_DIM * D_MODEL) / 1024, 256, 0, stream>>>(W2, W2H);

  // 1) LN1: x -> xn (f16)
  ln_kernel<<<ROWS, 256, 0, stream>>>(x, g1, be1, xn);
  // 2) q,k,v = xn @ W{q,k,v}^T + b -> [b][h][s][hd] f16
  qkv_gemm_kernel<<<dim3(D_MODEL / 128, ROWS / 128, 3), 256, 0, stream>>>(
      xn, WqH, bq, WkH, bk, WvH, bv, qB, kB, vB);
  // 3) causal flash attention -> ctx (f16, token-major)
  attn_kernel<<<dim3(SEQ / 64, BATCH * N_HEADS), 128, 0, stream>>>(qB, kB, vB,
                                                                   ctx);
  // 4) x2 = x + ctx @ Wo^T + bo (f32)
  gemm_kernel<0><<<dim3(D_MODEL / 128, ROWS / 128), 256, 0, stream>>>(
      ctx, WoH, bo, x, x2, nullptr, D_MODEL, D_MODEL);
  // 5) LN2: x2 -> xn (f16)
  ln_kernel<<<ROWS, 256, 0, stream>>>(x2, g2, be2, xn);
  // 6) h = gelu(xn @ W1^T + b1) (f16)
  gemm_kernel<1><<<dim3(FF_DIM / 128, ROWS / 128), 256, 0, stream>>>(
      xn, W1H, b1, nullptr, nullptr, hff, FF_DIM, D_MODEL);
  // 7) out = x2 + h @ W2^T + b2 (f32)
  gemm_kernel<0><<<dim3(D_MODEL / 128, ROWS / 128), 256, 0, stream>>>(
      hff, W2H, b2, x2, out, nullptr, D_MODEL, FF_DIM);
}
